// FP8Linear_47906065219950
// MI455X (gfx1250) — compile-verified, matrix operand order checked by reading, not run on hardware
//
#include <hip/hip_runtime.h>
#include <hip/hip_bf16.h>

// ---------------------------------------------------------------------------
// FP8 (e4m3) dynamic per-tensor quantized linear:  out = (xq*sx) @ (wq*sw)^T
// x: [M,K] bf16,  weight: [N,K] bf16,  out: [M,N] bf16.  M=32768, K=N=2048.
// GEMM uses v_wmma_f32_16x16x128_fp8_fp8 with double-buffered LDS staged via
// gfx1250 async global->LDS copies (ASYNCcnt).
// ---------------------------------------------------------------------------

typedef int    v16i __attribute__((ext_vector_type(16)));
typedef int    v4i  __attribute__((ext_vector_type(4)));
typedef float  v8f  __attribute__((ext_vector_type(8)));
typedef unsigned int v2u __attribute__((ext_vector_type(2)));

#define D_IN  2048
#define D_OUT 2048
#define FP8_MAX 448.0f

// ------------------------------ helpers -----------------------------------

__device__ __forceinline__ float bf16_bits_to_f32(unsigned hi16) {
    union { unsigned u; float f; } c;
    c.u = hi16 << 16;
    return c.f;
}

__device__ __forceinline__ unsigned short f32_to_bf16_rne(float f) {
    union { float f; unsigned u; } c;
    c.f = f;
    unsigned r = c.u + 0x7FFFu + ((c.u >> 16) & 1u);
    return (unsigned short)(r >> 16);
}

// Software f32 -> e4m3 (saturating, approx RNE). Fallback only.
__device__ __forceinline__ unsigned char f32_to_e4m3_sw(float f) {
    union { float f; unsigned u; } c;
    c.f = f;
    unsigned char s = (unsigned char)((c.u >> 24) & 0x80u);
    unsigned a = c.u & 0x7FFFFFFFu;
    if (a >= 0x43E00000u) return (unsigned char)(s | 0x7Eu);   // clamp to 448
    if (a < 0x3C000000u) {                                      // < 2^-7 -> denorm
        float scaled = __builtin_fabsf(f) * 512.0f;             // * 2^9
        int q = (int)(scaled + 0.5f);
        if (q > 7) q = 7;
        return (unsigned char)(s | (unsigned char)q);
    }
    unsigned mant = a & 0x7FFFFFu;
    unsigned exp  = a >> 23;
    unsigned r = mant + 0x7FFFFu + ((mant >> 20) & 1u);         // RNE at bit 20
    if (r & 0x800000u) { exp++; r = 0; }
    unsigned e8 = exp - 127u + 7u;
    return (unsigned char)(s | (unsigned char)((e8 << 3) | ((r >> 20) & 7u)));
}

__device__ __forceinline__ unsigned pack4_e4m3(float f0, float f1, float f2, float f3) {
#if __has_builtin(__builtin_amdgcn_cvt_pk_fp8_f32)
    int p = 0;
    p = __builtin_amdgcn_cvt_pk_fp8_f32(f0, f1, p, false);   // low word
    p = __builtin_amdgcn_cvt_pk_fp8_f32(f2, f3, p, true);    // high word
    return (unsigned)p;
#else
    return (unsigned)f32_to_e4m3_sw(f0)
         | ((unsigned)f32_to_e4m3_sw(f1) << 8)
         | ((unsigned)f32_to_e4m3_sw(f2) << 16)
         | ((unsigned)f32_to_e4m3_sw(f3) << 24);
#endif
}

// gfx1250 async global->LDS copy: per lane, 16B MEM[vaddr] -> LDS[lds_off].
// LDS byte address == low 32 bits of the generic shared-space pointer
__device__ __forceinline__ void async_copy_b128(const void* gptr, void* lds_ptr) {
    unsigned lds_off = (unsigned)(unsigned long long)(uintptr_t)lds_ptr;
    unsigned long long ga = (unsigned long long)(uintptr_t)gptr;
    asm volatile("global_load_async_to_lds_b128 %0, %1, off"
                 :
                 : "v"(lds_off), "v"(ga)
                 : "memory");
}

__device__ __forceinline__ void wait_asynccnt0() {
#if __has_builtin(__builtin_amdgcn_s_wait_asynccnt)
    __builtin_amdgcn_s_wait_asynccnt(0);
#else
    asm volatile("s_wait_asynccnt 0x0" ::: "memory");
#endif
}

// ------------------------------ kernels -----------------------------------

__global__ void init_ws_kernel(unsigned* amax_bits) {
    amax_bits[0] = 0u;
    amax_bits[1] = 0u;
}

// abs(bf16) bit-pattern max is monotone in value -> integer atomicMax works.
__global__ void amax_bf16_kernel(const unsigned* __restrict__ in,  // 2 bf16 / uint
                                 long n_pairs, unsigned* __restrict__ out_bits) {
    __shared__ unsigned red[256];
    unsigned m = 0u;
    long stride = (long)gridDim.x * blockDim.x;
    for (long i = (long)blockIdx.x * blockDim.x + threadIdx.x; i < n_pairs; i += stride) {
        unsigned v  = in[i];
        unsigned lo = v & 0x7FFFu;
        unsigned hi = (v >> 16) & 0x7FFFu;
        unsigned t  = lo > hi ? lo : hi;
        m = t > m ? t : m;
    }
    red[threadIdx.x] = m;
    __syncthreads();
    for (int s = 128; s > 0; s >>= 1) {
        if ((int)threadIdx.x < s) {
            unsigned o = red[threadIdx.x + s];
            if (o > red[threadIdx.x]) red[threadIdx.x] = o;
        }
        __syncthreads();
    }
    if (threadIdx.x == 0) atomicMax(out_bits, red[0]);
}

// scales[0] = sx_dequant, scales[1] = sw_dequant, scales[2] = sx_q, scales[3] = sw_q
__global__ void compute_scales_kernel(const unsigned* __restrict__ amax_bits,
                                      float* __restrict__ scales) {
    union { unsigned u; float f; } cx, cw;
    cx.u = amax_bits[0] << 16;
    cw.u = amax_bits[1] << 16;
    float ax = fmaxf(cx.f, 1e-12f);
    float aw = fmaxf(cw.f, 1e-12f);
    scales[0] = ax * (1.0f / FP8_MAX);
    scales[1] = aw * (1.0f / FP8_MAX);
    scales[2] = FP8_MAX / ax;
    scales[3] = FP8_MAX / aw;
}

// 4 bf16 -> 4 fp8 per thread-iteration.
__global__ void quantize_e4m3_kernel(const unsigned* __restrict__ in,   // 2 bf16 / uint
                                     unsigned* __restrict__ outq,       // 4 fp8 / uint
                                     long n_quads,
                                     const float* __restrict__ scales, int sidx) {
    const float qs = scales[sidx];
    long stride = (long)gridDim.x * blockDim.x;
    for (long i = (long)blockIdx.x * blockDim.x + threadIdx.x; i < n_quads; i += stride) {
        v2u v = ((const v2u*)in)[i];
        float f0 = bf16_bits_to_f32(v.x & 0xFFFFu)         * qs;
        float f1 = bf16_bits_to_f32((v.x >> 16) & 0xFFFFu) * qs;
        float f2 = bf16_bits_to_f32(v.y & 0xFFFFu)         * qs;
        float f3 = bf16_bits_to_f32((v.y >> 16) & 0xFFFFu) * qs;
        outq[i] = pack4_e4m3(f0, f1, f2, f3);
    }
}

// ------------------------------ FP8 GEMM ----------------------------------
// Block tile 128x128, K-step 128, double-buffered LDS staged asynchronously.
// 256 threads = 8 waves (4 along M, 2 along N); each wave: 2x4 tiles of 16x16
// -> 8x v_wmma_f32_16x16x128_fp8_fp8 per K-step.

#define BM 128
#define BN 128
#define BK 128
#define LDB (BK + 16)   // padded LDS row stride (bytes)

union FragAB { v16i v; v4i q[4]; };

__global__ void __launch_bounds__(256)
fp8_gemm_nt_kernel(const unsigned char* __restrict__ Aq,   // [M,K] fp8
                   const unsigned char* __restrict__ Bq,   // [N,K] fp8
                   unsigned short* __restrict__ out,       // [M,N] bf16 bits
                   const float* __restrict__ scales,
                   int M, int N, int K) {
    __shared__ __align__(16) unsigned char a_lds[2][BM * LDB];
    __shared__ __align__(16) unsigned char b_lds[2][BN * LDB];

    const int tid  = threadIdx.x;
    const int lane = tid & 31;
    const int wave = tid >> 5;      // 0..7
    const int wm   = wave & 3;      // wave row   (4 along M)
    const int wn   = wave >> 2;     // wave col   (2 along N)
    const int bm   = blockIdx.y * BM;
    const int bn   = blockIdx.x * BN;

    const float s = scales[0] * scales[1];   // combined dequant scale

    v8f acc[2][4];
#pragma unroll
    for (int i = 0; i < 2; ++i)
#pragma unroll
        for (int j = 0; j < 4; ++j)
            acc[i][j] = (v8f)0.0f;

    const int r    = lane & 15;
    const int half = lane >> 4;          // 0: K 0..63, 1: K 64..127 of the row

    // This thread's staging assignment: 4 chunks of A + 4 chunks of B per tile.
    const int srow = tid >> 1;           // 0..127: row handled (four 16B chunks per thread)
    const int sch0 = (tid & 1) * 4;      // chunk 0..3 or 4..7

    // Prologue: stage tile 0 into buffer 0.
#pragma unroll
    for (int c = 0; c < 4; ++c) {
        int ch = sch0 + c;
        async_copy_b128(Aq + (size_t)(bm + srow) * K + ch * 16,
                        &a_lds[0][srow * LDB + ch * 16]);
        async_copy_b128(Bq + (size_t)(bn + srow) * K + ch * 16,
                        &b_lds[0][srow * LDB + ch * 16]);
    }

    const int KT = K / BK;               // 16
    for (int kt = 0; kt < KT; ++kt) {
        const int buf = kt & 1;

        // Tile kt resident in LDS; previous buffer free for reuse.
        wait_asynccnt0();
        __syncthreads();

        // Kick off the next tile's async copies into the other buffer.
        if (kt + 1 < KT) {
            const int nbuf = buf ^ 1;
            const int k1 = (kt + 1) * BK;
#pragma unroll
            for (int c = 0; c < 4; ++c) {
                int ch = sch0 + c;
                async_copy_b128(Aq + (size_t)(bm + srow) * K + k1 + ch * 16,
                                &a_lds[nbuf][srow * LDB + ch * 16]);
                async_copy_b128(Bq + (size_t)(bn + srow) * K + k1 + ch * 16,
                                &b_lds[nbuf][srow * LDB + ch * 16]);
            }
        }

        // ---- fragment loads: lane -> matrix row, contiguous-K bytes ----
        FragAB af[2], bfr[4];
#pragma unroll
        for (int tm = 0; tm < 2; ++tm) {
            int row = wm * 32 + tm * 16 + r;
#pragma unroll
            for (int i = 0; i < 4; ++i)
                afr_load: af[tm].q[i] =
                    *(const v4i*)(&a_lds[buf][row * LDB + half * 64 + i * 16]);
        }
#pragma unroll
        for (int tn = 0; tn < 4; ++tn) {
            int row = wn * 64 + tn * 16 + r;
#pragma unroll
            for (int i = 0; i < 4; ++i)
                bfr[tn].q[i] =
                    *(const v4i*)(&b_lds[buf][row * LDB + half * 64 + i * 16]);
        }

        // ---- 8 x WMMA 16x16x128 fp8 (overlaps with async copies) ----
#pragma unroll
        for (int tm = 0; tm < 2; ++tm)
#pragma unroll
            for (int tn = 0; tn < 4; ++tn)
                acc[tm][tn] = __builtin_amdgcn_wmma_f32_16x16x128_fp8_fp8(
                    af[tm].v, bfr[tn].v, (short)0, acc[tm][tn], false, false);
    }

    // ---- epilogue: scale, cvt to bf16, store ----
    // C/D layout: VGPR v -> rows (v, v+8); lanes 0..15 cols, lanes 16..31 rows +8.
#pragma unroll
    for (int tm = 0; tm < 2; ++tm) {
#pragma unroll
        for (int tn = 0; tn < 4; ++tn) {
            int col   = bn + wn * 64 + tn * 16 + r;
            int mbase = bm + wm * 32 + tm * 16 + half * 8;
#pragma unroll
            for (int v = 0; v < 8; ++v) {
                float f = acc[tm][tn][v] * s;
                out[(size_t)(mbase + v) * N + col] = f32_to_bf16_rne(f);
            }
        }
    }
}

// ------------------------------ launch ------------------------------------

extern "C" void kernel_launch(void* const* d_in, const int* in_sizes, int n_in,
                              void* d_out, int out_size, void* d_ws, size_t ws_size,
                              hipStream_t stream) {
    const unsigned* x_bits = (const unsigned*)d_in[0];   // [M, K] bf16 (2 / uint)
    const unsigned* w_bits = (const unsigned*)d_in[1];   // [N, K] bf16
    unsigned short* out    = (unsigned short*)d_out;     // [M, N] bf16 bits

    const int K = D_IN;
    const int N = D_OUT;
    const int M = in_sizes[0] / K;                       // 32768

    // workspace layout
    unsigned*      amax_bits = (unsigned*)d_ws;                          // [2]
    float*         scales    = (float*)((char*)d_ws + 16);               // [4]
    unsigned char* xq        = (unsigned char*)d_ws + 256;               // M*K fp8
    unsigned char* wq        = xq + (size_t)M * K;                       // N*K fp8

    // 1) zero amax accumulators (ws is not re-poisoned between replays)
    init_ws_kernel<<<1, 1, 0, stream>>>(amax_bits);

    // 2) per-tensor amax
    long x_pairs = (long)M * K / 2;
    long w_pairs = (long)N * K / 2;
    amax_bf16_kernel<<<2048, 256, 0, stream>>>(x_bits, x_pairs, amax_bits + 0);
    amax_bf16_kernel<<<512,  256, 0, stream>>>(w_bits, w_pairs, amax_bits + 1);

    // 3) scales
    compute_scales_kernel<<<1, 1, 0, stream>>>(amax_bits, scales);

    // 4) quantize to fp8 e4m3
    long x_quads = (long)M * K / 4;
    long w_quads = (long)N * K / 4;
    quantize_e4m3_kernel<<<4096, 256, 0, stream>>>(x_bits, (unsigned*)xq, x_quads, scales, 2);
    quantize_e4m3_kernel<<<1024, 256, 0, stream>>>(w_bits, (unsigned*)wq, w_quads, scales, 3);

    // 5) FP8 WMMA GEMM, bf16 output
    dim3 grid(N / BN, M / BM);
    fp8_gemm_nt_kernel<<<grid, 256, 0, stream>>>(xq, wq, out, scales, M, N, K);
}
